// FP8Linear_12360915878778
// MI455X (gfx1250) — 8-bit matrix operand order repaired and checked by reading, compiled, not run on hardware
//
#include <hip/hip_runtime.h>

typedef __attribute__((ext_vector_type(16))) int      v16i;
typedef __attribute__((ext_vector_type(8)))  float    v8f;
typedef __attribute__((ext_vector_type(8)))  _Float16 v8h;
typedef __attribute__((ext_vector_type(4)))  unsigned v4u;
typedef __attribute__((ext_vector_type(8)))  unsigned v8u;
typedef unsigned long long u64;

#define TILE_K   128
#define LDS_PAD  16
#define LDS_ROW  (TILE_K + LDS_PAD)
#define BLK_M    128
#define BLK_N    256

__global__ __launch_bounds__(256)
void fp8_quant_kernel(const _Float16* __restrict__ in,
                      unsigned char*  __restrict__ outq,
                      const _Float16* __restrict__ scale,
                      long n)
{
    long i = ((long)blockIdx.x * blockDim.x + threadIdx.x) * 8;
    if (i >= n) return;
    const float inv = 1.0f / (float)scale[0];

    v8h x = __builtin_nontemporal_load((const v8h*)(in + i));
    float f[8];
#pragma unroll
    for (int j = 0; j < 8; ++j) {
        float y = (float)x[j] * inv;
        f[j] = fminf(fmaxf(y, -448.0f), 448.0f);
    }
    int lo = __builtin_amdgcn_cvt_pk_fp8_f32(f[0], f[1], 0,  false);
    lo     = __builtin_amdgcn_cvt_pk_fp8_f32(f[2], f[3], lo, true);
    int hi = __builtin_amdgcn_cvt_pk_fp8_f32(f[4], f[5], 0,  false);
    hi     = __builtin_amdgcn_cvt_pk_fp8_f32(f[6], f[7], hi, true);

    *(u64*)(outq + i) = ((u64)(unsigned)hi << 32) | (unsigned)lo;
}

__device__ __forceinline__ void tdm_load_tile_2d(unsigned ldsOff,
                                                 const void* gaddr,
                                                 unsigned strideElems,
                                                 unsigned tileRows)
{
    u64 ga = (u64)(__SIZE_TYPE__)gaddr;
    v4u g0;
    g0[0] = 1u;
    g0[1] = ldsOff;
    g0[2] = (unsigned)ga;
    g0[3] = (unsigned)((ga >> 32) & 0x01ffffffu)
            | (2u << 30);

    const unsigned td0 = 1u << 20;
    const unsigned td1 = 1u << 20;
    v8u g1;
    g1[0] = (1u << 20)
          | (4u << 22)
          | (3u << 25);
    g1[1] = (td0 & 0xffffu) << 16;
    g1[2] = (td0 >> 16) | ((td1 & 0xffffu) << 16);
    g1[3] = (td1 >> 16) | ((unsigned)TILE_K << 16);
    g1[4] = tileRows;
    g1[5] = strideElems;
    g1[6] = 0u;
    g1[7] = 0u;
    asm volatile("tensor_load_to_lds %0, %1" :: "s"(g0), "s"(g1) : "memory");
}

__global__ __launch_bounds__(256)
void fp8_gemm_kernel(const unsigned char* __restrict__ Aq,
                     const unsigned char* __restrict__ Wq,
                     const _Float16*      __restrict__ s_in,
                     const _Float16*      __restrict__ s_w,
                     _Float16*            __restrict__ out,
                     int M, int N, int K)
{
    __shared__ __align__(16) unsigned char sA[2][BLK_M * LDS_ROW];
    __shared__ __align__(16) unsigned char sB[2][BLK_N * LDS_ROW];

    const int lane    = threadIdx.x & 31;
    const int wave    = threadIdx.x >> 5;
    const int waveM   = wave >> 2;
    const int waveN   = wave & 3;
    const int laneLow = lane & 15;
    const int laneHi  = lane >> 4;

    const long blockM = (long)blockIdx.y * BLK_M;
    const long blockN = (long)blockIdx.x * BLK_N;

    v8f acc[4][4];
#pragma unroll
    for (int t = 0; t < 4; ++t)
#pragma unroll
        for (int u = 0; u < 4; ++u)
#pragma unroll
            for (int r = 0; r < 8; ++r)
                acc[t][u][r] = 0.0f;

    unsigned ldsA[2], ldsB[2];
    ldsA[0] = (unsigned)(__SIZE_TYPE__)&sA[0][0];
    ldsA[1] = (unsigned)(__SIZE_TYPE__)&sA[1][0];
    ldsB[0] = (unsigned)(__SIZE_TYPE__)&sB[0][0];
    ldsB[1] = (unsigned)(__SIZE_TYPE__)&sB[1][0];

    int aOff[4], bOff[4];
#pragma unroll
    for (int t = 0; t < 4; ++t)
        aOff[t] = (waveM * 64 + 16 * t + laneLow) * LDS_ROW + 16 * laneHi;
#pragma unroll
    for (int u = 0; u < 4; ++u)
        bOff[u] = (waveN * 64 + 16 * u + laneLow) * LDS_ROW + 16 * laneHi;

    const unsigned char* aTileG = Aq + blockM * (long)K;
    const unsigned char* bTileG = Wq + blockN * (long)K;
    const int nSteps = K >> 7;

    if (wave == 0) tdm_load_tile_2d(ldsA[0], aTileG, (unsigned)K, BLK_M);
    if (wave == 1) tdm_load_tile_2d(ldsB[0], bTileG, (unsigned)K, BLK_N);
    __builtin_amdgcn_s_wait_tensorcnt(0);
    __syncthreads();

    for (int ks = 0; ks < nSteps; ++ks) {
        const int cur = ks & 1;

        if (ks + 1 < nSteps) {
            if (wave == 0)
                tdm_load_tile_2d(ldsA[cur ^ 1], aTileG + (long)(ks + 1) * TILE_K,
                                 (unsigned)K, BLK_M);
            if (wave == 1)
                tdm_load_tile_2d(ldsB[cur ^ 1], bTileG + (long)(ks + 1) * TILE_K,
                                 (unsigned)K, BLK_N);
        }

        const unsigned char* tA = &sA[cur][0];
        const unsigned char* tB = &sB[cur][0];

        v16i b[4];
#pragma unroll
        for (int u = 0; u < 4; ++u)
#pragma unroll
            for (int g = 0; g < 4; ++g) {
                int4 d = *(const int4*)(tB + bOff[u] + 32 * g);
                b[u][4 * g + 0] = d.x;
                b[u][4 * g + 1] = d.y;
                b[u][4 * g + 2] = d.z;
                b[u][4 * g + 3] = d.w;
            }

#pragma unroll
        for (int t = 0; t < 4; ++t) {
            v16i a;
#pragma unroll
            for (int h = 0; h < 2; ++h)
#pragma unroll
                for (int c = 0; c < 4; ++c) {
                    u64 d = *(const u64*)(tA + aOff[t] + 64 * h + 32 * (c >> 1) + 8 * (c & 1));
                    a[8 * h + 2 * c + 0] = (int)(unsigned)(d & 0xffffffffull);
                    a[8 * h + 2 * c + 1] = (int)(unsigned)(d >> 32);
                }
#pragma unroll
            for (int u = 0; u < 4; ++u)
                acc[t][u] = __builtin_amdgcn_wmma_f32_16x16x128_fp8_fp8(
                    a, b[u], (short)0, acc[t][u], false, false);
        }

        __builtin_amdgcn_s_wait_tensorcnt(0);
        __syncthreads();
    }

    const float s = (float)s_in[0] * (float)s_w[0];
#pragma unroll
    for (int t = 0; t < 4; ++t)
#pragma unroll
        for (int u = 0; u < 4; ++u) {
            const long col = blockN + waveN * 64 + 16 * u + laneLow;
#pragma unroll
            for (int r = 0; r < 8; ++r) {
                const long row = blockM + waveM * 64 + 16 * t + r + 8 * laneHi;
                __builtin_nontemporal_store((_Float16)(acc[t][u][r] * s),
                                            out + row * (long)N + col);
            }
        }
}

extern "C" void kernel_launch(void* const* d_in, const int* in_sizes, int n_in,
                              void* d_out, int out_size, void* d_ws, size_t ws_size,
                              hipStream_t stream)
{
    const _Float16* X    = (const _Float16*)d_in[0];
    const _Float16* W    = (const _Float16*)d_in[1];
    const _Float16* s_in = (const _Float16*)d_in[2];
    const _Float16* s_w  = (const _Float16*)d_in[3];
    _Float16*       out  = (_Float16*)d_out;

    const long K = 4096;
    const long M = (long)in_sizes[0] / K;
    const long N = (long)in_sizes[1] / K;

    unsigned char* Xq = (unsigned char*)d_ws;
    unsigned char* Wq = Xq + (size_t)M * K;

    {
        long n = M * K;
        long threads = n / 8;
        dim3 grid((unsigned)((threads + 255) / 256));
        fp8_quant_kernel<<<grid, 256, 0, stream>>>(X, Xq, s_in, n);
    }
    {
        long n = N * K;
        long threads = n / 8;
        dim3 grid((unsigned)((threads + 255) / 256));
        fp8_quant_kernel<<<grid, 256, 0, stream>>>(W, Wq, s_w, n);
    }
    {
        dim3 grid((unsigned)(N / BLK_N), (unsigned)(M / BLK_M));
        fp8_gemm_kernel<<<grid, 256, 0, stream>>>(Xq, Wq, s_in, s_w, out,
                                                  (int)M, (int)N, (int)K);
    }
}
